// MinkUNet_52209622450769
// MI455X (gfx1250) — matrix operand order repaired and checked by reading, not compiled, not run on hardware
//
#include <hip/hip_runtime.h>
#include <hip/hip_bf16.h>

// MinkUNet (dense 32^3 grid) forward on gfx1250.
//
// Conv = implicit GEMM on v_wmma_f32_16x16x32_f16 with 2x2 register tiling:
// one wave owns a 32-voxel x 32-outchannel tile (4 f32 accumulators).
// Steady-state inner loop: 8x global_load_b128 (2 A-frags + 2 B-frags)
// feeding 4x v_wmma -> 2 loads per WMMA (vs 4 for a 16x16 tile).
// Weights are pre-transposed/converted to f16 [tap][CoutP][CinP]; invalid
// rows/taps read a zeroed workspace page (pointer select, no branches).
// Training-mode BatchNorm: per-channel reduction + fused
// normalize/affine/(skip)/(ReLU) epilogue writing f16.
//
// Assumptions (compile-only environment):
// - d_in leaves follow JAX pytree flattening (dict keys sorted at every
//   level): cls_b, cls_w, s1..s4, stem, u1..u4, then x.
// - B-fragment lane layout matches A's (AMD's dense 16-bit table order)
//   (N = lane&15; lane half h holds K 8h..8h+7, then K 16+8h..23+8h).
// - ws_size >= ~75 MB.

typedef _Float16 h16;
typedef __attribute__((ext_vector_type(8)))  _Float16 h16x8;
typedef __attribute__((ext_vector_type(16))) _Float16 v16h;
typedef __attribute__((ext_vector_type(8)))  float    v8f;

// Build a 16-element fragment from two contiguous 8-element runs.
__device__ __forceinline__ v16h ldfrag(const h16* p, int off2)
{
    h16x8 x0 = *(const h16x8*)(p);
    h16x8 x1 = *(const h16x8*)(p + off2);
    return __builtin_shufflevector(x0, x1,
        0,1,2,3,4,5,6,7,8,9,10,11,12,13,14,15);
}

// ---------------------------------------------------------------------------
// Branchless implicit-GEMM conv, 32x32 output tile per wave.
//  mode 0: normal conv (K, stride, pad).
//  mode 1: transposed 2x2x2 stride-2 conv: out[o] uses src=o>>1, tap=1-(o&1).
// Requirements: Cin % 32 == 0 (X row stride = Cin), Cout % 32 == 0
// (Wh has Cout rows per tap, O row stride = Cout), O has ceil(N/32)*32 rows capacity,
// ZPAD is a zeroed region of >= Cin+32 f16.
// ---------------------------------------------------------------------------
__global__ __launch_bounds__(32)
void conv3d_wmma_kernel(const h16* __restrict__ X, const h16* __restrict__ Wh,
                        const h16* __restrict__ ZPAD, float* __restrict__ O,
                        int Din, int Hin, int Win, int Cin,
                        int Dout, int Hout, int Wout, int Cout,
                        int K, int stride, int pad, int mode)
{
    const int lane  = threadIdx.x;
    const int half  = lane >> 4;
    const int halfo = half << 3;            // 8*half: A k-offset AND D row offset
    const int ml    = lane & 15;            // A row / B,D column (within 16-tile)
    const int m0    = blockIdx.x << 5;      // 32 voxels per wave
    const int n0    = blockIdx.y << 5;      // 32 out channels per wave
    const int Nvox  = Dout * Hout * Wout;

    // Two A-row voxels per lane (M-subtiles 0 and 1).
    int  od[2], oh[2], ow[2];
    bool mv[2];
#pragma unroll
    for (int s = 0; s < 2; ++s) {
        const int vox = m0 + (s << 4) + ml;
        mv[s] = vox < Nvox;
        const int vv = mv[s] ? vox : 0;
        od[s] = vv / (Hout * Wout);
        const int rr = vv - od[s] * Hout * Wout;
        oh[s] = rr / Wout;
        ow[s] = rr - oh[s] * Wout;
    }

    const int co0 = n0 + ml;                // N-subtile 0 column
    const int kcN = Cin >> 5;
    const int T   = K * K * K;

    const long wtap       = (long)Cin * Cout;
    const long nsub       = (long)Cin << 4; // 16 rows of Wh between N-subtiles
    const h16* const pbt0 = Wh + (long)co0 * Cin + (half << 3);

    v8f acc00 = {}, acc01 = {}, acc10 = {}, acc11 = {};

    for (int t = 0; t < T; ++t) {
        const int td = t / (K * K);
        const int r2 = t - td * K * K;
        const int th = r2 / K;
        const int tw = r2 - th * K;

        // Per-M-subtile source pointer (ZPAD when out of range / tap mismatch).
        const h16* pa[2];
#pragma unroll
        for (int s = 0; s < 2; ++s) {
            bool ok;
            int id, ih, iw;
            if (mode == 0) {
                id = od[s] * stride + td - pad;
                ih = oh[s] * stride + th - pad;
                iw = ow[s] * stride + tw - pad;
                ok = mv[s] && (unsigned)id < (unsigned)Din &&
                              (unsigned)ih < (unsigned)Hin &&
                              (unsigned)iw < (unsigned)Win;
            } else {                        // transposed k=2 s=2 VALID
                ok = mv[s] && (td == ((od[s] & 1) ^ 1)) &&
                              (th == ((oh[s] & 1) ^ 1)) &&
                              (tw == ((ow[s] & 1) ^ 1));
                id = od[s] >> 1; ih = oh[s] >> 1; iw = ow[s] >> 1;
            }
            pa[s] = ok ? (X + (long)((id * Hin + ih) * Win + iw) * Cin + halfo)
                       : (ZPAD + halfo);
        }
        const h16* pb = pbt0 + (long)t * wtap;

        for (int kc = 0; kc < kcN; ++kc) {
            const int ko = kc << 5;
            v16h a0 = ldfrag(pa[0] + ko, 16);
            v16h a1 = ldfrag(pa[1] + ko, 16);
            v16h b0 = ldfrag(pb + ko, 16);
            v16h b1 = ldfrag(pb + nsub + ko, 16);
            acc00 = __builtin_amdgcn_wmma_f32_16x16x32_f16(
                        false, a0, false, b0, (short)0, acc00, false, false);
            acc01 = __builtin_amdgcn_wmma_f32_16x16x32_f16(
                        false, a0, false, b1, (short)0, acc01, false, false);
            acc10 = __builtin_amdgcn_wmma_f32_16x16x32_f16(
                        false, a1, false, b0, (short)0, acc10, false, false);
            acc11 = __builtin_amdgcn_wmma_f32_16x16x32_f16(
                        false, a1, false, b1, (short)0, acc11, false, false);
        }
    }

    // D store: within a 16x16 tile, row r maps to M = r + 8*half, col = ml.
    // Unconditional (output buffers are padded to whole 32-row tiles).
#pragma unroll
    for (int r = 0; r < 8; ++r) {
        const long ra = (long)(m0 + r + halfo) * Cout;
        const long rb = (long)(m0 + 16 + r + halfo) * Cout;
        O[ra + co0]      = acc00[r];
        O[ra + co0 + 16] = acc01[r];
        O[rb + co0]      = acc10[r];
        O[rb + co0 + 16] = acc11[r];
    }
}

// ---------------------------------------------------------------------------
// Weight prep: f32 [T][Cin][Cout] -> f16 [T][CoutP][CinP], zero padded.
// ---------------------------------------------------------------------------
__global__ __launch_bounds__(256)
void wprep_kernel(const float* __restrict__ W, h16* __restrict__ Wh,
                  int T, int Cin, int Cout, int CinP, int CoutP)
{
    long i = (long)blockIdx.x * 256 + threadIdx.x;
    long total = (long)T * CoutP * CinP;
    if (i >= total) return;
    int  ci = (int)(i % CinP);
    long r  = i / CinP;
    int  co = (int)(r % CoutP);
    int  t  = (int)(r / CoutP);
    float v = (ci < Cin && co < Cout) ? W[((long)t * Cin + ci) * Cout + co] : 0.f;
    Wh[i] = (h16)v;
}

__global__ __launch_bounds__(256)
void zero_h16_kernel(h16* __restrict__ p, long n)
{
    long i = (long)blockIdx.x * 256 + threadIdx.x;
    if (i < n) p[i] = (h16)0.f;
}

// Pad f32 [N][Cin] -> f16 [N][CP] (zeros beyond Cin).
__global__ __launch_bounds__(256)
void xpad_kernel(const float* __restrict__ x, h16* __restrict__ out,
                 long N, int Cin, int CP)
{
    long i = (long)blockIdx.x * 256 + threadIdx.x;
    if (i >= N * CP) return;
    long v = i / CP;
    int  c = (int)(i - v * CP);
    out[i] = (c < Cin) ? (h16)x[v * Cin + c] : (h16)0.f;
}

// ---------------------------------------------------------------------------
// BatchNorm statistics: one block per channel.
// ---------------------------------------------------------------------------
__global__ __launch_bounds__(256)
void bn_stats_kernel(const float* __restrict__ X, int N, int C,
                     float* __restrict__ mean, float* __restrict__ rstd)
{
    const int c = blockIdx.x;
    float s = 0.f, s2 = 0.f;
    for (int v = threadIdx.x; v < N; v += 256) {
        float x = X[(long)v * C + c];
        s += x; s2 += x * x;
    }
    __shared__ float sh[256], sh2[256];
    sh[threadIdx.x] = s; sh2[threadIdx.x] = s2;
    __syncthreads();
    for (int o = 128; o > 0; o >>= 1) {
        if (threadIdx.x < o) {
            sh[threadIdx.x]  += sh[threadIdx.x + o];
            sh2[threadIdx.x] += sh2[threadIdx.x + o];
        }
        __syncthreads();
    }
    if (threadIdx.x == 0) {
        float m = sh[0] / (float)N;
        float v = sh2[0] / (float)N - m * m;
        mean[c] = m;
        rstd[c] = rsqrtf(v + 1e-5f);
    }
}

// BN apply: y = (x-mean)*rstd*g + b  [+ skip] [ReLU] -> f16 (strided dest).
__global__ __launch_bounds__(256)
void bn_apply_kernel(const float* __restrict__ X,
                     const float* __restrict__ mean, const float* __restrict__ rstd,
                     const float* __restrict__ g, const float* __restrict__ bb,
                     const h16* __restrict__ skip, h16* __restrict__ Out,
                     int N, int C, int outStride, int outOfs, int relu)
{
    long i = (long)blockIdx.x * 256 + threadIdx.x;
    if (i >= (long)N * C) return;
    long v = i / C;
    int  c = (int)(i - v * C);
    float y = (X[i] - mean[c]) * rstd[c] * g[c] + bb[c];
    if (skip) y += (float)skip[v * C + c];
    if (relu) y = fmaxf(y, 0.f);
    Out[v * outStride + outOfs + c] = (h16)y;
}

__global__ __launch_bounds__(256)
void copy_strided_kernel(const h16* __restrict__ src, h16* __restrict__ dst,
                         long n, int C, int dstStride, int dstOfs)
{
    long i = (long)blockIdx.x * 256 + threadIdx.x;
    if (i >= n) return;
    long v = i / C;
    int  c = (int)(i - v * C);
    dst[v * dstStride + dstOfs + c] = src[i];
}

// Final epilogue: out[v*C+c] = F[v*CP+c] + bias[c]  (f32).
__global__ __launch_bounds__(256)
void out_bias_kernel(const float* __restrict__ F, const float* __restrict__ bias,
                     float* __restrict__ out, long N, int C, int CP)
{
    long i = (long)blockIdx.x * 256 + threadIdx.x;
    if (i >= N * C) return;
    long v = i / C;
    int  c = (int)(i - v * C);
    out[i] = F[v * CP + c] + bias[c];
}

// ---------------------------------------------------------------------------
// Host-side parameter unpacking (JAX pytree sorted-key order).
// ---------------------------------------------------------------------------
struct BNP   { const float *b, *g; };
struct ResP  { BNP bn1, bn2, bnd; const float *w1, *w2, *wd; };
struct StageP{ BNP dbn; const float* dw; ResP r1, r2; };

static BNP getbn(void* const* d, int& i)
{ BNP p; p.b = (const float*)d[i++]; p.g = (const float*)d[i++]; return p; }

static ResP getres(void* const* d, int& i, bool wd)
{
    ResP p{}; p.bn1 = getbn(d, i); p.bn2 = getbn(d, i);
    if (wd) p.bnd = getbn(d, i);
    p.w1 = (const float*)d[i++];
    p.w2 = (const float*)d[i++];
    p.wd = wd ? (const float*)d[i++] : nullptr;
    return p;
}

static StageP getstage(void* const* d, int& i, bool wd)
{
    StageP s; s.dbn = getbn(d, i); s.dw = (const float*)d[i++];
    s.r1 = getres(d, i, wd); s.r2 = getres(d, i, false);
    return s;
}

// ---------------------------------------------------------------------------
// Host launch helpers.
// ---------------------------------------------------------------------------
struct Ctx {
    hipStream_t st;
    h16*  WBUF;                 // padded/transposed f16 weights (reused)
    const h16* ZPAD;            // zeroed page for invalid A rows
    float *F0, *F1, *mean, *rstd;
};

// Prep weights then run conv. CinL may be < X's row stride only for the
// stem (X pre-padded); CinP is X's row stride (mult of 32), CoutP mult of 32.
static void conv_pc(const Ctx& c, const h16* X, const float* Wf, float* O,
                    int Din, int Hin, int Win, int CinL,
                    int Dout, int Hout, int Wout, int CoutL,
                    int K, int stride, int pad, int mode)
{
    const int CinP  = (CinL + 31) & ~31;
    const int CoutP = (CoutL + 31) & ~31;
    const int T = K * K * K;
    long wtot = (long)T * CoutP * CinP;
    wprep_kernel<<<(int)((wtot + 255) / 256), 256, 0, c.st>>>(
        Wf, c.WBUF, T, CinL, CoutL, CinP, CoutP);
    dim3 grid((Dout * Hout * Wout + 31) / 32, CoutP / 32);
    conv3d_wmma_kernel<<<grid, dim3(32), 0, c.st>>>(X, c.WBUF, c.ZPAD, O,
        Din, Hin, Win, CinP, Dout, Hout, Wout, CoutP, K, stride, pad, mode);
}

static void bnrun(const Ctx& c, const float* F, int N, int C, BNP p,
                  const h16* skip, h16* out, int ostr, int oofs, int relu)
{
    bn_stats_kernel<<<C, 256, 0, c.st>>>(F, N, C, c.mean, c.rstd);
    long tot = (long)N * C;
    bn_apply_kernel<<<(int)((tot + 255) / 256), 256, 0, c.st>>>(
        F, c.mean, c.rstd, p.g, p.b, skip, out, N, C, ostr, oofs, relu);
}

static void res_block(const Ctx& c, int D, int H, int W,
                      const h16* xin, int Cin, int Cout, const ResP& p,
                      h16* t1, h16* t2, h16* out)
{
    const int N = D * H * W;
    conv_pc(c, xin, p.w1, c.F0, D, H, W, Cin, D, H, W, Cout, 3, 1, 1, 0);
    bnrun(c, c.F0, N, Cout, p.bn1, nullptr, t1, Cout, 0, 1);
    conv_pc(c, t1, p.w2, c.F1, D, H, W, Cout, D, H, W, Cout, 3, 1, 1, 0);
    const h16* sc = xin;
    if (p.wd) {
        conv_pc(c, xin, p.wd, c.F0, D, H, W, Cin, D, H, W, Cout, 1, 1, 0, 0);
        bnrun(c, c.F0, N, Cout, p.bnd, nullptr, t2, Cout, 0, 0);
        sc = t2;
    }
    bnrun(c, c.F1, N, Cout, p.bn2, sc, out, Cout, 0, 1);
}

extern "C" void kernel_launch(void* const* d_in, const int* in_sizes, int n_in,
                              void* d_out, int out_size, void* d_ws, size_t ws_size,
                              hipStream_t stream)
{
    (void)in_sizes; (void)n_in; (void)out_size; (void)ws_size;

    // ---- unpack params (sorted pytree leaf order) ----
    int i = 0;
    const float* cls_b = (const float*)d_in[i++];
    const float* cls_w = (const float*)d_in[i++];
    StageP s1 = getstage(d_in, i, false);
    StageP s2 = getstage(d_in, i, true);
    StageP s3 = getstage(d_in, i, true);
    StageP s4 = getstage(d_in, i, true);
    ResP   stem = getres(d_in, i, false);   // keys bn1,bn2,w1,w2 == res w/o wd
    StageP u1 = getstage(d_in, i, true);
    StageP u2 = getstage(d_in, i, true);
    StageP u3 = getstage(d_in, i, true);
    StageP u4 = getstage(d_in, i, true);
    const float* x = (const float*)d_in[i++];

    // ---- workspace carve (bump allocator, 256B aligned) ----
    char* wp = (char*)d_ws;
    auto alloc = [&](size_t bytes) -> void* {
        void* p = wp; wp += (bytes + 255) & ~(size_t)255; return p;
    };
    const long V32 = 32768, V16 = 4096, V8 = 512, V4 = 64;
    const long ZN   = 512;                                  // zero page (f16)
    const long WMAX = 27L * 384 * 256;                      // largest padded W
    h16*  ZPAD = (h16*)alloc(ZN * sizeof(h16));
    h16*  WBUF = (h16*)alloc(WMAX * sizeof(h16));
    h16*  HX = (h16*)alloc(V32 * 32 * sizeof(h16));         // x padded 4->32
    h16*  X0 = (h16*)alloc(V32 * 32 * sizeof(h16));         // skip buffers
    h16*  X1 = (h16*)alloc(V16 * 32 * sizeof(h16));
    h16*  X2 = (h16*)alloc(V8  * 64 * sizeof(h16));
    h16*  X3 = (h16*)alloc(V4  * 128 * sizeof(h16));
    h16*  H0 = (h16*)alloc(V32 * 128 * sizeof(h16));        // generic f16 pool
    h16*  H1 = (h16*)alloc(V32 * 128 * sizeof(h16));
    h16*  H2 = (h16*)alloc(V32 * 128 * sizeof(h16));
    h16*  H3 = (h16*)alloc(V32 * 128 * sizeof(h16));
    float* F0 = (float*)alloc(V32 * 96 * sizeof(float));    // conv f32 outputs
    float* F1 = (float*)alloc(V32 * 96 * sizeof(float));
    float* mean = (float*)alloc(512 * sizeof(float));
    float* rstd = (float*)alloc(512 * sizeof(float));

    Ctx c{stream, WBUF, ZPAD, F0, F1, mean, rstd};

    zero_h16_kernel<<<(int)((ZN + 255) / 256), 256, 0, stream>>>(ZPAD, ZN);

    // ---- stem (32^3, 4 -> 32 -> 32) ----
    xpad_kernel<<<(int)((V32 * 32 + 255) / 256), 256, 0, stream>>>(x, HX, V32, 4, 32);
    conv_pc(c, HX, stem.w1, F0, 32,32,32,4, 32,32,32,32, 3,1,1,0);
    bnrun(c, F0, (int)V32, 32, stem.bn1, nullptr, H0, 32, 0, 1);
    conv_pc(c, H0, stem.w2, F0, 32,32,32,32, 32,32,32,32, 3,1,1,0);
    bnrun(c, F0, (int)V32, 32, stem.bn2, nullptr, X0, 32, 0, 1);

    // ---- encoder: stride-2 conv + res x2 ----
    auto down = [&](const StageP& sp, const h16* xin, int Dinn,
                    int Cin, int Cout, h16* out) {
        const int Do = Dinn / 2, N = Do * Do * Do;
        conv_pc(c, xin, sp.dw, F0, Dinn,Dinn,Dinn,Cin, Do,Do,Do,Cin, 2,2,0,0);
        bnrun(c, F0, N, Cin, sp.dbn, nullptr, H0, Cin, 0, 1);
        res_block(c, Do,Do,Do, H0, Cin,  Cout, sp.r1, H1, H2, H3);
        res_block(c, Do,Do,Do, H3, Cout, Cout, sp.r2, H1, H2, out);
    };
    down(s1, X0, 32, 32,  32,  X1);
    down(s2, X1, 16, 32,  64,  X2);
    down(s3, X2, 8,  64,  128, X3);
    down(s4, X3, 4,  128, 256, H0);      // x4 -> H0  [8, 256]

    // ---- decoder: transposed conv + concat skip + res x2 ----
    // A=xin (dead after convT), B=cat then stage-out, C=mid, D=tmp.
    auto up = [&](const StageP& upp, h16* A, h16* B, h16* Cm, h16* Dv,
                  int Dinn, int Cin, int Cout, const h16* skip, int Cskip) {
        const int Do = Dinn * 2, N = Do * Do * Do, Ct = Cout + Cskip;
        conv_pc(c, A, upp.dw, F0, Dinn,Dinn,Dinn,Cin, Do,Do,Do,Cout, 2,2,0,1);
        bnrun(c, F0, N, Cout, upp.dbn, nullptr, B, Ct, 0, 1);
        copy_strided_kernel<<<(int)(((long)N * Cskip + 255) / 256), 256, 0, stream>>>(
            skip, B, (long)N * Cskip, Cskip, Ct, Cout);
        res_block(c, Do,Do,Do, B, Ct,   Cout, upp.r1, Dv, A, Cm);
        res_block(c, Do,Do,Do, Cm, Cout, Cout, upp.r2, Dv, A, B);
    };
    up(u1, H0, H1, H2, H3, 2,  256, 256, X3, 128);   // y1 -> H1 [64, 256]
    up(u2, H1, H0, H2, H3, 4,  256, 128, X2, 64);    // y2 -> H0 [512, 128]
    up(u3, H0, H1, H2, H3, 8,  128, 96,  X1, 32);    // y3 -> H1 [4096, 96]
    up(u4, H1, H0, H2, H3, 16, 96,  96,  X0, 32);    // y4 -> H0 [32768, 96]

    // ---- classifier: 1x1x1 conv (96 -> 20 padded to 32) + bias ----
    conv_pc(c, H0, cls_w, F0, 32,32,32,96, 32,32,32,20, 1,1,0,0);
    out_bias_kernel<<<(int)((V32 * 20 + 255) / 256), 256, 0, stream>>>(
        F0, cls_b, (float*)d_out, V32, 20, 32);
}
